// m_GCN_257698037954
// MI455X (gfx1250) — matrix operand order repaired and checked by reading, not compiled, not run on hardware
//
#include <hip/hip_runtime.h>
#include <hip/hip_bf16.h>
#include <math.h>

#define N_NODES 10000
#define N_EDGES 160000
#define IN_DIM  3
#define LATENT  96
#define HIDDEN  192
#define NLAYER  45
#define GEN_EPS 1e-7f

// WMMA fragment types (wave32, 16x16x32 f16 -> f32)
typedef _Float16 v16h __attribute__((ext_vector_type(16)));
typedef float    v8f  __attribute__((ext_vector_type(8)));
// TDM descriptor group types (probe-verified builtin signatures)
typedef unsigned int u32x4 __attribute__((ext_vector_type(4)));
typedef int          i32x4 __attribute__((ext_vector_type(4)));
typedef int          i32x8 __attribute__((ext_vector_type(8)));

#define TILES_PER_BLOCK 4                  // 2 waves * 2 tiles
#define N_TILES     ((N_NODES + 15) / 16)                          // 625
#define MLP_BLOCKS  ((N_TILES + TILES_PER_BLOCK - 1) / TILES_PER_BLOCK)   // 157
#define N_TILES_PAD (MLP_BLOCKS * TILES_PER_BLOCK)                 // 628
#define N_ROWS_PAD  (N_TILES_PAD * 16)                             // 10048

#define W1_NT 12                   // 192/16 column tiles of W1
#define W1_KF 3                    // 96/32 K fragments
#define W2_NT 6                    // 96/16 column tiles of W2
#define W2_KF 6                    // 192/32 K fragments
#define W1_HALVES (W1_NT * W1_KF * 32 * 16)   // 18432 halves = 36864 B
#define W2_HALVES (W2_NT * W2_KF * 32 * 16)   // 18432 halves = 36864 B
#define LAYER_HALVES (W1_HALVES + W2_HALVES)  // 36864 halves = 73728 B

// ---------------------------------------------------------------------------
// One-time: re-pack W1/W2 (f32 row-major) into f16 WMMA B-fragment layout,
// layer-contiguous; a layer's W1 and W2 are each staged by one TDM descriptor.
// B layout (32x16 f16): lane holds column n = lane&15; lanes 0-15 hold
// K = kbase+0..7 (elems 0..7), kbase+16..23 (elems 8..15); lanes 16-31 hold K = kbase+8..15, kbase+24..31.
// wpack layout: [layer]( [nt][kf][lane][16] for W1 | [ot][kf][lane][16] for W2 )
// ---------------------------------------------------------------------------
__global__ void pack_w1_kernel(const float* __restrict__ w1, _Float16* __restrict__ wp) {
    int t = blockIdx.x * blockDim.x + threadIdx.x;
    const int total = NLAYER * W1_NT * W1_KF * 32;
    if (t >= total) return;
    int lane  = t & 31;
    int kf    = (t >> 5) % W1_KF;
    int nt    = (t / (32 * W1_KF)) % W1_NT;
    int layer = t / (32 * W1_KF * W1_NT);
    int col   = nt * 16 + (lane & 15);
    int kbase = kf * 32 + ((lane >> 4) * 8);
    const float* src = w1 + (size_t)layer * LATENT * HIDDEN;
    _Float16* dst = wp + (size_t)layer * LAYER_HALVES
                       + ((size_t)((nt * W1_KF + kf) * 32 + lane)) * 16;
#pragma unroll
    for (int j = 0; j < 16; ++j)
        dst[j] = (_Float16)src[(size_t)(kbase + j + (j & 8)) * HIDDEN + col];
}

__global__ void pack_w2_kernel(const float* __restrict__ w2, _Float16* __restrict__ wp) {
    int t = blockIdx.x * blockDim.x + threadIdx.x;
    const int total = NLAYER * W2_NT * W2_KF * 32;
    if (t >= total) return;
    int lane  = t & 31;
    int kf    = (t >> 5) % W2_KF;
    int nt    = (t / (32 * W2_KF)) % W2_NT;
    int layer = t / (32 * W2_KF * W2_NT);
    int col   = nt * 16 + (lane & 15);
    int kbase = kf * 32 + ((lane >> 4) * 8);
    const float* src = w2 + (size_t)layer * HIDDEN * LATENT;
    _Float16* dst = wp + (size_t)layer * LAYER_HALVES + W1_HALVES
                       + ((size_t)((nt * W2_KF + kf) * 32 + lane)) * 16;
#pragma unroll
    for (int j = 0; j < 16; ++j)
        dst[j] = (_Float16)src[(size_t)(kbase + j + (j & 8)) * LATENT + col];
}

// ---------------------------------------------------------------------------
// z0 = x @ w_node_in ; also seed h = z0. Pad rows [N_NODES, N_ROWS_PAD) are
// zero-seeded so the padded MLP tiles never see poison.
// ---------------------------------------------------------------------------
__global__ void node_embed_kernel(const float* __restrict__ x,
                                  const float* __restrict__ w_in,
                                  float* __restrict__ z, float* __restrict__ h) {
    int i = blockIdx.x * blockDim.x + threadIdx.x;
    if (i >= N_ROWS_PAD * LATENT) return;
    int n = i / LATENT, l = i - n * LATENT;
    float v = 0.f;
    if (n < N_NODES) {
        const float* xr = x + (size_t)n * IN_DIM;
        v = fmaf(xr[0], w_in[l],
            fmaf(xr[1], w_in[LATENT + l],
                 xr[2] * w_in[2 * LATENT + l]));
    }
    z[i] = v;
    h[i] = v;
}

// ---------------------------------------------------------------------------
// Message + aggregate: h[dst] += relu(z[src] + edge_attr @ w_edge) + EPS
// ---------------------------------------------------------------------------
__global__ void msg_aggr_kernel(const float* __restrict__ z,
                                const int* __restrict__ ei,
                                const float* __restrict__ ea,
                                const float* __restrict__ w_edge,
                                float* __restrict__ h) {
    int t = blockIdx.x * blockDim.x + threadIdx.x;
    if (t >= N_EDGES * (LATENT / 4)) return;
    int e = t / (LATENT / 4);
    int c = (t - e * (LATENT / 4)) * 4;
    int src = ei[e];
    int dst = ei[N_EDGES + e];
    float a0 = ea[(size_t)e * 2 + 0];
    float a1 = ea[(size_t)e * 2 + 1];
    float4 zv = *(const float4*)(z + (size_t)src * LATENT + c);
    float4 w0 = *(const float4*)(w_edge + c);
    float4 w1 = *(const float4*)(w_edge + LATENT + c);
    float m0 = fmaxf(zv.x + fmaf(a0, w0.x, a1 * w1.x), 0.f) + GEN_EPS;
    float m1 = fmaxf(zv.y + fmaf(a0, w0.y, a1 * w1.y), 0.f) + GEN_EPS;
    float m2 = fmaxf(zv.z + fmaf(a0, w0.z, a1 * w1.z), 0.f) + GEN_EPS;
    float m3 = fmaxf(zv.w + fmaf(a0, w0.w, a1 * w1.w), 0.f) + GEN_EPS;
    float* hp = h + (size_t)dst * LATENT + c;
    atomicAdd(hp + 0, m0);
    atomicAdd(hp + 1, m1);
    atomicAdd(hp + 2, m2);
    atomicAdd(hp + 3, m3);
}

// ---------------------------------------------------------------------------
// TDM: 1-D tile DMA of `bytes` (multiple of 8) from global into LDS.
// D# per CDNA5 ISA.
// ---------------------------------------------------------------------------
#if __has_builtin(__builtin_amdgcn_tensor_load_to_lds)
__device__ __forceinline__ void tdm_load_1d(const void* gsrc, unsigned lds_off,
                                            unsigned bytes) {
    unsigned units = bytes >> 3;                 // 8-byte elements
    unsigned long long ga = (unsigned long long)(uintptr_t)gsrc;
    u32x4 g0;
    g0[0] = 1u;                                  // count=1, user-mode load D#
    g0[1] = lds_off;                             // lds_addr (bytes)
    g0[2] = (unsigned)ga;                        // global_addr[31:0]
    g0[3] = (unsigned)((ga >> 32) & 0x01FFFFFFu) | (2u << 30);  // addr[56:32]|type=2
    i32x8 g1;
    g1[0] = 3 << 16;                             // data_size = 8 bytes
    g1[1] = (int)((units & 0xFFFFu) << 16);      // tensor_dim0[15:0] (bits 63:48)
    g1[2] = (int)(((units >> 16) & 0xFFFFu) | (1u << 16)); // dim0 hi | tensor_dim1=1
    g1[3] = (int)((units & 0xFFFFu) << 16);      // tile_dim0 (bits 127:112)
    g1[4] = 1;                                   // tile_dim1 = 1
    g1[5] = (int)units;                          // tensor_dim0_stride lo32
    g1[6] = 0;
    g1[7] = 0;
    i32x4 gz = {0, 0, 0, 0};
#if __clang_major__ >= 23
    i32x8 gz8 = {0, 0, 0, 0, 0, 0, 0, 0};
    __builtin_amdgcn_tensor_load_to_lds(g0, g1, gz, gz, gz8, 0);
#else
    __builtin_amdgcn_tensor_load_to_lds(g0, g1, gz, gz, 0);
#endif
}
#endif

// Stage `bytes` of weights into LDS, visible to the whole block on return.
__device__ __forceinline__ void stage_weights(const _Float16* gsrc,
                                              _Float16* ldst, unsigned bytes) {
#if __has_builtin(__builtin_amdgcn_tensor_load_to_lds)
    if (threadIdx.x < 32) {                      // wave 0 issues the DMA
        tdm_load_1d(gsrc, (unsigned)(uintptr_t)ldst, bytes);
        __builtin_amdgcn_s_wait_tensorcnt(0);
    }
#else
    for (unsigned off = threadIdx.x * 16u; off < bytes; off += blockDim.x * 16u)
        *(uint4*)((char*)ldst + off) = *(const uint4*)((const char*)gsrc + off);
#endif
    __syncthreads();
}

// ---------------------------------------------------------------------------
// Fused MLP layer: z_next = selu( relu(h @ W1) @ W2 ); h_next = z_next.
// 64 threads = 2 waves; each wave owns TWO 16-row tiles so every LDS-resident
// weight fragment feeds 2 WMMAs (144 v_wmma per wave). Layer weights are
// TDM-DMA'd into LDS (W1 for phase 1, then W2 reuses the same region).
// Grid covers padded rows -> NO per-tile guards: every wave fully active,
// EXEC statically all-ones around WMMA, branch-free inner loops.
// LDS: 36864 B weights + 2 waves * 2 tiles * 16x192 f16 staging = 60 KB.
// ---------------------------------------------------------------------------
__global__ __launch_bounds__(64) void mlp_wmma_kernel(
        const float* __restrict__ hbuf,
        const _Float16* __restrict__ wpack,
        float* __restrict__ z, float* __restrict__ hout, int layer) {
    __shared__ __align__(16) _Float16 smem_w[W1_HALVES];          // 36864 B
    __shared__ __align__(16) _Float16 smem_c[2][2][16 * HIDDEN];  // 24576 B
    const int wave = threadIdx.x >> 5;
    const int lane = threadIdx.x & 31;
    const int m  = lane & 15;       // A row within tile == C column index
    const int hi = lane >> 4;
    const int t0 = blockIdx.x * TILES_PER_BLOCK + wave * 2;
    const int t1 = t0 + 1;
    const _Float16* wl = wpack + (size_t)layer * LAYER_HALVES;

    // ---- stage W1 fragments into LDS (TDM) ----
    stage_weights(wl, smem_w, W1_HALVES * 2);

    // ---- GEMM1: relu(h @ W1) for both tiles, sharing B fragments ----
    // A layout (16x32 f16): lane<16: elems 0..7 -> K+0..7, 8..15 -> K+16..23;
    //                       lane>=16: elems 0..7 -> K+8..15, 8..15 -> K+24..31.
    v16h a0[W1_KF], a1[W1_KF];
    {
        const float* hr0 = hbuf + (size_t)(t0 * 16 + m) * LATENT;
        const float* hr1 = hbuf + (size_t)(t1 * 16 + m) * LATENT;
#pragma unroll
        for (int kf = 0; kf < W1_KF; ++kf) {
            int k0 = kf * 32 + hi * 8;
#pragma unroll
            for (int j = 0; j < 8; ++j) a0[kf][j]     = (_Float16)hr0[k0 + j];
#pragma unroll
            for (int j = 0; j < 8; ++j) a0[kf][8 + j] = (_Float16)hr0[k0 + 16 + j];
#pragma unroll
            for (int j = 0; j < 8; ++j) a1[kf][j]     = (_Float16)hr1[k0 + j];
#pragma unroll
            for (int j = 0; j < 8; ++j) a1[kf][8 + j] = (_Float16)hr1[k0 + 16 + j];
        }
    }
#pragma unroll
    for (int nt = 0; nt < W1_NT; ++nt) {
        v8f acc0 = {}, acc1 = {};
#pragma unroll
        for (int kf = 0; kf < W1_KF; ++kf) {
            v16h b = *(const v16h*)(smem_w + ((nt * W1_KF + kf) * 32 + lane) * 16);
            acc0 = __builtin_amdgcn_wmma_f32_16x16x32_f16(
                       false, a0[kf], false, b, (short)0, acc0, false, false);
            acc1 = __builtin_amdgcn_wmma_f32_16x16x32_f16(
                       false, a1[kf], false, b, (short)0, acc1, false, false);
        }
        // C layout: lane holds col n=lane&15, rows hi*8..hi*8+7 in elems 0..7.
#pragma unroll
        for (int r = 0; r < 8; ++r) {
            smem_c[wave][0][(hi * 8 + r) * HIDDEN + nt * 16 + m] =
                (_Float16)fmaxf(acc0[r], 0.f);
            smem_c[wave][1][(hi * 8 + r) * HIDDEN + nt * 16 + m] =
                (_Float16)fmaxf(acc1[r], 0.f);
        }
    }
    __syncthreads();                     // phase-1 reads of smem_w complete

    // ---- stage W2 fragments (reuse the same LDS region, TDM again) ----
    stage_weights(wl + W1_HALVES, smem_w, W2_HALVES * 2);

    // ---- GEMM2: (.) @ W2, selu epilogue, both tiles ----
    v16h a2_0[W2_KF], a2_1[W2_KF];
#pragma unroll
    for (int kf = 0; kf < W2_KF; ++kf) {
        int k0 = kf * 32 + hi * 8;
#pragma unroll
        for (int j = 0; j < 8; ++j) a2_0[kf][j]     = smem_c[wave][0][m * HIDDEN + k0 + j];
#pragma unroll
        for (int j = 0; j < 8; ++j) a2_0[kf][8 + j] = smem_c[wave][0][m * HIDDEN + k0 + 16 + j];
#pragma unroll
        for (int j = 0; j < 8; ++j) a2_1[kf][j]     = smem_c[wave][1][m * HIDDEN + k0 + j];
#pragma unroll
        for (int j = 0; j < 8; ++j) a2_1[kf][8 + j] = smem_c[wave][1][m * HIDDEN + k0 + 16 + j];
    }
    const float selu_scale = 1.0507009873554805f;
    const float selu_sa    = 1.0507009873554805f * 1.6732632423543772f;
#pragma unroll
    for (int ot = 0; ot < W2_NT; ++ot) {
        v8f acc0 = {}, acc1 = {};
#pragma unroll
        for (int kf = 0; kf < W2_KF; ++kf) {
            v16h b = *(const v16h*)(smem_w + ((ot * W2_KF + kf) * 32 + lane) * 16);
            acc0 = __builtin_amdgcn_wmma_f32_16x16x32_f16(
                       false, a2_0[kf], false, b, (short)0, acc0, false, false);
            acc1 = __builtin_amdgcn_wmma_f32_16x16x32_f16(
                       false, a2_1[kf], false, b, (short)0, acc1, false, false);
        }
        int colg = ot * 16 + m;
        int rowb0 = t0 * 16 + hi * 8;
        int rowb1 = t1 * 16 + hi * 8;
#pragma unroll
        for (int r = 0; r < 8; ++r) {
            float v0 = acc0[r];
            v0 = (v0 > 0.f) ? selu_scale * v0 : selu_sa * (__expf(v0) - 1.f);
            size_t i0 = (size_t)(rowb0 + r) * LATENT + colg;
            z[i0] = v0;
            hout[i0] = v0;               // residual base for next layer
            float v1 = acc1[r];
            v1 = (v1 > 0.f) ? selu_scale * v1 : selu_sa * (__expf(v1) - 1.f);
            size_t i1 = (size_t)(rowb1 + r) * LATENT + colg;
            z[i1] = v1;
            hout[i1] = v1;
        }
    }
}

// ---------------------------------------------------------------------------
// out = concat(y, z @ w_node_out)
// ---------------------------------------------------------------------------
__global__ void predict_kernel(const float* __restrict__ z,
                               const float* __restrict__ w_out,
                               const float* __restrict__ y,
                               float* __restrict__ out) {
    int n = blockIdx.x * blockDim.x + threadIdx.x;
    if (n >= N_NODES) return;
    const float* zr = z + (size_t)n * LATENT;
    float s = 0.f;
#pragma unroll
    for (int l = 0; l < LATENT; ++l) s = fmaf(zr[l], w_out[l], s);
    out[n] = y[n];
    out[N_NODES + n] = s;
}

// ---------------------------------------------------------------------------
extern "C" void kernel_launch(void* const* d_in, const int* in_sizes, int n_in,
                              void* d_out, int out_size, void* d_ws, size_t ws_size,
                              hipStream_t stream) {
    const float* x          = (const float*)d_in[0];
    const float* y          = (const float*)d_in[1];
    const float* edge_attr  = (const float*)d_in[2];
    const float* w_node_in  = (const float*)d_in[3];
    const float* w_edge     = (const float*)d_in[4];
    const float* w1         = (const float*)d_in[5];
    const float* w2         = (const float*)d_in[6];
    const float* w_node_out = (const float*)d_in[7];
    const int*   edge_index = (const int*)d_in[8];
    float* out = (float*)d_out;

    // workspace: z | h (padded to whole blocks) | wpack(f16)   (~11 MB, L2-resident)
    float* z = (float*)d_ws;
    float* h = z + (size_t)N_ROWS_PAD * LATENT;
    _Float16* wpack = (_Float16*)(h + (size_t)N_ROWS_PAD * LATENT);

    // One-time f16 fragment pack of all 45 layers' weights.
    {
        int t1 = NLAYER * W1_NT * W1_KF * 32;
        int t2 = NLAYER * W2_NT * W2_KF * 32;
        pack_w1_kernel<<<(t1 + 255) / 256, 256, 0, stream>>>(w1, wpack);
        pack_w2_kernel<<<(t2 + 255) / 256, 256, 0, stream>>>(w2, wpack);
    }

    // Input embedding (also seeds residual buffer h and zeroes pad rows).
    node_embed_kernel<<<(N_ROWS_PAD * LATENT + 255) / 256, 256, 0, stream>>>(
        x, w_node_in, z, h);

    // 45 GENConv layers (stream-ordered; all state L2-resident).
    const int msg_threads = N_EDGES * (LATENT / 4);
    for (int L = 0; L < NLAYER; ++L) {
        msg_aggr_kernel<<<(msg_threads + 255) / 256, 256, 0, stream>>>(
            z, edge_index, edge_attr, w_edge, h);
        mlp_wmma_kernel<<<MLP_BLOCKS, 64, 0, stream>>>(h, wpack, z, h, L);
    }

    predict_kernel<<<(N_NODES + 255) / 256, 256, 0, stream>>>(
        z, w_node_out, y, out);
}
